// JiTBlock_90769838833769
// MI455X (gfx1250) — matrix operand order repaired and checked by reading, not compiled, not run on hardware
//
#include <hip/hip_runtime.h>
#include <hip/hip_bf16.h>
#include <math.h>

typedef __attribute__((ext_vector_type(16))) __bf16 v16bf;
typedef __attribute__((ext_vector_type(8)))  float  v8f;

#define DEV static __device__ __forceinline__
#define HD 64

union FragU { v16bf v; uint4 q[2]; };

// A-matrix 16x32 bf16 fragment: lane m=lane&15, half=lane>>4.
// VGPR0-3 hold K = 8*half .. 8*half+7 ; VGPR4-7 hold K = 16+8*half .. +7.
DEV v16bf load_frag_a(const __bf16* rowbase, int half) {
  FragU f;
  f.q[0] = *(const uint4*)(rowbase + 8 * half);
  f.q[1] = *(const uint4*)(rowbase + 16 + 8 * half);
  return f.v;
}
// B-matrix 32x16 bf16 fragment from LDS laid out [N][K]: lane n=lane&15.
// VGPR0-3 hold K = 8*half .. 8*half+7 ; VGPR4-7 hold K = 16+8*half .. +7.
DEV v16bf load_frag_b(const __bf16* rowbase, int half) {
  FragU f;
  f.q[0] = *(const uint4*)(rowbase + 8 * half);
  f.q[1] = *(const uint4*)(rowbase + 16 + 8 * half);
  return f.v;
}
DEV v8f wmma_bf16(v16bf a, v16bf b, v8f c) {
  return __builtin_amdgcn_wmma_f32_16x16x32_bf16(false, a, false, b, (short)0, c,
                                                 false, false);
}
DEV float silu_f(float x) { return x / (1.f + __expf(-x)); }

// Async global->LDS 16-byte copy (CDNA5, tracked by ASYNCcnt, bypasses VGPRs).
// Generic shared pointers carry the LDS byte address in their low 32 bits.
DEV void async_b128(void* lds_dst, const void* gsrc) {
  unsigned loff = (unsigned)(unsigned long long)lds_dst;
  asm volatile("global_load_async_to_lds_b128 %0, %1, off"
               :: "v"(loff), "v"((unsigned long long)gsrc)
               : "memory");
}
DEV void wait_async0() {
  asm volatile("s_wait_asynccnt 0x0" ::: "memory");
}

// ---------------------------------------------------------------------------
// Generic bf16 WMMA GEMM:  C[M,N] = A[M,K] * BT[N,K]^T + bias
// mode 0: outf = C+bias (fp32)
// mode 1: outb = C+bias (bf16)
// mode 2: outf = X + mods[b*modstride+goff+col] * (C+bias)   (fused residual)
// Double-buffered async global->LDS staging for full tiles; element-wise
// guarded fallback for ragged N blocks and the K remainder.
// ---------------------------------------------------------------------------
__global__ __launch_bounds__(256)
void gemm_kernel(const __bf16* __restrict__ A, const __bf16* __restrict__ BT,
                 const float* __restrict__ bias,
                 int M, int N, int K, int mode,
                 float* __restrict__ outf, __bf16* __restrict__ outb,
                 const float* __restrict__ X, const float* __restrict__ mods,
                 int goff, int modstride, int S)
{
  __shared__ __bf16 As[2][128][40];  // 32 cols + 8 pad (80B stride, 16B aligned)
  __shared__ __bf16 Bs[2][128][40];  // [n][k]
  const int tid = threadIdx.x;
  const int lane = tid & 31, wave = tid >> 5;
  const int wm = wave & 3, wn = wave >> 2;          // 4x2 wave grid
  const int bm = blockIdx.y * 128, bn = blockIdx.x * 128;
  const int half = lane >> 4, nl = lane & 15;

  v8f acc[2][4] = {};

  auto issue_tile = [&](int buf, int kt) {
    for (int i = tid; i < 512; i += 256) {
      int r = i >> 2, cg = (i & 3) << 3;
      async_b128(&As[buf][r][cg], A + (size_t)(bm + r) * K + kt + cg);
      async_b128(&Bs[buf][r][cg], BT + (size_t)(bn + r) * K + kt + cg);
    }
  };
  auto stage_elem = [&](int buf, int kt) {
    for (int i = tid; i < 4096; i += 256) {
      int r = i >> 5, cc = i & 31;
      int gk = kt + cc;
      As[buf][r][cc] = (gk < K) ? A[(size_t)(bm + r) * K + gk] : (__bf16)0.f;
      __bf16 bv = (__bf16)0.f;
      if (bn + r < N && gk < K) bv = BT[(size_t)(bn + r) * K + gk];
      Bs[buf][r][cc] = bv;
    }
  };
  auto compute_tile = [&](int buf) {
    v16bf af[2], bfr[4];
    af[0] = load_frag_a(&As[buf][wm * 32 + nl][0], half);
    af[1] = load_frag_a(&As[buf][wm * 32 + 16 + nl][0], half);
    for (int nt = 0; nt < 4; ++nt)
      bfr[nt] = load_frag_b(&Bs[buf][wn * 64 + nt * 16 + nl][0], half);
    for (int mt = 0; mt < 2; ++mt)
      for (int nt = 0; nt < 4; ++nt)
        acc[mt][nt] = wmma_bf16(af[mt], bfr[nt], acc[mt][nt]);
  };

  const int nfull = K >> 5;
  const int tail  = K & 31;
  const bool asyncok = (bn + 128 <= N);   // async copies cannot zero-fill

  if (asyncok && nfull > 0) {
    issue_tile(0, 0);
    for (int j = 0; j < nfull; ++j) {
      int cur = j & 1;
      wait_async0();          // own copies for tile j complete
      __syncthreads();        // everyone's copies done; prior-buffer reads done
      if (j + 1 < nfull) issue_tile(cur ^ 1, (j + 1) * 32);
      compute_tile(cur);
    }
    if (tail) {
      __syncthreads();
      stage_elem(nfull & 1, nfull * 32);
      __syncthreads();
      compute_tile(nfull & 1);
    }
  } else {
    for (int kt = 0; kt < K; kt += 32) {
      stage_elem(0, kt);
      __syncthreads();
      compute_tile(0);
      __syncthreads();
    }
  }

  for (int mt = 0; mt < 2; ++mt) {
    for (int nt = 0; nt < 4; ++nt) {
      int col = bn + wn * 64 + nt * 16 + nl;
      if (col >= N) continue;
      float bv = bias[col];
      for (int r = 0; r < 8; ++r) {
        int row = bm + wm * 32 + mt * 16 + half * 8 + r;   // C layout: m=r+8*half
        float cv = acc[mt][nt][r] + bv;
        size_t idx = (size_t)row * N + col;
        if (mode == 0) outf[idx] = cv;
        else if (mode == 1) outb[idx] = (__bf16)cv;
        else {
          int b = row / S;
          outf[idx] = X[idx] + mods[(size_t)b * modstride + goff + col] * cv;
        }
      }
    }
  }
}

// ---------------------------------------------------------------------------
// Flash-style attention: one (b,h) per blockIdx.y, 64 queries per block,
// 4 waves x 16 query rows. q,k: [B*H][S][64] bf16; vt: [B*H][64][S] bf16.
// Double-buffered async K/V staging; o: [B][S][H][64] bf16.
// ---------------------------------------------------------------------------
__global__ __launch_bounds__(128)
void attn_kernel(const __bf16* __restrict__ q, const __bf16* __restrict__ k,
                 const __bf16* __restrict__ vt, __bf16* __restrict__ o,
                 int Sn, int Hn)
{
  __shared__ __bf16 Ks[2][64][72];   // [key][hd]
  __shared__ __bf16 Vs[2][64][72];   // [hd][key]  (== V^T, B-fragment ready)
  __shared__ __bf16 Ps[4][16][72];   // per-wave P staging [qrow][key]
  const int tid = threadIdx.x, lane = tid & 31, w = tid >> 5;
  const int half = lane >> 4, nl = lane & 15;
  const int bh = blockIdx.y;
  const int b = bh / Hn, h = bh % Hn;
  const int q0 = blockIdx.x * 64 + w * 16;

  auto issue_chunk = [&](int buf, int kc) {
    for (int i = tid; i < 512; i += 128) {
      int r = i >> 3, cg = (i & 7) << 3;
      async_b128(&Ks[buf][r][cg], k + ((size_t)bh * Sn + kc + r) * HD + cg);
      async_b128(&Vs[buf][r][cg], vt + ((size_t)bh * HD + r) * Sn + kc + cg);
    }
  };

  const __bf16* qrow = q + ((size_t)bh * Sn + q0 + nl) * HD;
  v16bf qa[2];
  qa[0] = load_frag_a(qrow, half);
  qa[1] = load_frag_a(qrow + 32, half);

  float rmax[8], rsum[8];
  for (int r = 0; r < 8; ++r) { rmax[r] = -3.0e38f; rsum[r] = 0.f; }
  v8f oacc[4] = {};

  const int nc = Sn / 64;
  issue_chunk(0, 0);
  for (int j = 0; j < nc; ++j) {
    int cur = j & 1;
    wait_async0();
    __syncthreads();
    if (j + 1 < nc) issue_chunk(cur ^ 1, (j + 1) * 64);

    v8f st[4];
    for (int t = 0; t < 4; ++t) {
      v8f s = {};
      for (int ks = 0; ks < 2; ++ks) {
        v16bf bfr = load_frag_b(&Ks[cur][t * 16 + nl][ks * 32], half);
        s = wmma_bf16(qa[ks], bfr, s);
      }
      for (int r = 0; r < 8; ++r) s[r] *= 0.125f;   // 1/sqrt(64)
      st[t] = s;
    }
    // online softmax: row m = 8*half + r lives in one VGPR across 16 lanes
    for (int r = 0; r < 8; ++r) {
      float m = fmaxf(fmaxf(st[0][r], st[1][r]), fmaxf(st[2][r], st[3][r]));
      for (int off = 8; off; off >>= 1) m = fmaxf(m, __shfl_xor(m, off, 32));
      float nm = fmaxf(rmax[r], m);
      float alpha = __expf(rmax[r] - nm);
      float psum = 0.f;
      for (int t = 0; t < 4; ++t) {
        float p = __expf(st[t][r] - nm);
        st[t][r] = p;
        psum += p;
      }
      for (int off = 8; off; off >>= 1) psum += __shfl_xor(psum, off, 32);
      rmax[r] = nm;
      rsum[r] = rsum[r] * alpha + psum;
      for (int t = 0; t < 4; ++t) oacc[t][r] *= alpha;
      for (int t = 0; t < 4; ++t)
        Ps[w][half * 8 + r][t * 16 + nl] = (__bf16)st[t][r];
    }
    // re-fragment P as WMMA A (same-wave LDS ops are in-order)
    v16bf pa[2];
    pa[0] = load_frag_a(&Ps[w][nl][0], half);
    pa[1] = load_frag_a(&Ps[w][nl][32], half);
    for (int t = 0; t < 4; ++t)
      for (int ks = 0; ks < 2; ++ks) {
        v16bf bv = load_frag_b(&Vs[cur][t * 16 + nl][ks * 32], half);
        oacc[t] = wmma_bf16(pa[ks], bv, oacc[t]);
      }
  }

  for (int r = 0; r < 8; ++r) {
    float inv = 1.f / rsum[r];
    int srow = q0 + half * 8 + r;
    for (int t = 0; t < 4; ++t) {
      int d = t * 16 + nl;
      o[(((size_t)b * Sn + srow) * Hn + h) * HD + d] = (__bf16)(oacc[t][r] * inv);
    }
  }
}

// ---------------------------------------------------------------------------
// qkv fp32 [B,S,3,H,64] -> per-head RMSNorm + RoPE -> q,k bf16 [B*H][S][64],
// v bf16 transposed [B*H][64][S]. One wave per (b,s,h); lane owns dims 2l,2l+1.
// ---------------------------------------------------------------------------
__global__ __launch_bounds__(256)
void rope_kernel(const float* __restrict__ qkv,
                 const float* __restrict__ cosb, const float* __restrict__ sinb,
                 const float* __restrict__ qs, const float* __restrict__ ksc,
                 __bf16* __restrict__ qo, __bf16* __restrict__ ko,
                 __bf16* __restrict__ vto, int Bn, int Sn, int Hn)
{
  int gw = blockIdx.x * 8 + (threadIdx.x >> 5);
  int lane = threadIdx.x & 31;
  int h = gw % Hn;
  int s = (gw / Hn) % Sn;
  int b = gw / (Hn * Sn);
  if (b >= Bn) return;
  int d0 = lane * 2, d1 = d0 + 1;
  size_t base = (((size_t)(b * Sn + s) * 3) * Hn + h) * HD;
  size_t bh = (size_t)(b * Hn + h);
  float c0 = cosb[s * HD + d0], c1 = cosb[s * HD + d1];
  float s0 = sinb[s * HD + d0], s1 = sinb[s * HD + d1];

  { // Q
    float x0 = qkv[base + d0], x1 = qkv[base + d1];
    float ss = x0 * x0 + x1 * x1;
    for (int off = 16; off; off >>= 1) ss += __shfl_xor(ss, off, 32);
    float rn = rsqrtf(ss * (1.f / HD) + 1e-6f);
    float y0 = x0 * rn * qs[d0], y1 = x1 * rn * qs[d1];
    qo[(bh * Sn + s) * HD + d0] = (__bf16)(y0 * c0 - y1 * s0);
    qo[(bh * Sn + s) * HD + d1] = (__bf16)(y1 * c1 + y0 * s1);
  }
  { // K
    size_t kb = base + (size_t)Hn * HD;
    float x0 = qkv[kb + d0], x1 = qkv[kb + d1];
    float ss = x0 * x0 + x1 * x1;
    for (int off = 16; off; off >>= 1) ss += __shfl_xor(ss, off, 32);
    float rn = rsqrtf(ss * (1.f / HD) + 1e-6f);
    float y0 = x0 * rn * ksc[d0], y1 = x1 * rn * ksc[d1];
    ko[(bh * Sn + s) * HD + d0] = (__bf16)(y0 * c0 - y1 * s0);
    ko[(bh * Sn + s) * HD + d1] = (__bf16)(y1 * c1 + y0 * s1);
  }
  { // V (store transposed)
    size_t vb = base + 2 * (size_t)Hn * HD;
    vto[(bh * HD + d0) * Sn + s] = (__bf16)qkv[vb + d0];
    vto[(bh * HD + d1) * Sn + s] = (__bf16)qkv[vb + d1];
  }
}

// RMSNorm over D + modulate -> bf16. One block (256 thr) per token row, D=1024.
__global__ __launch_bounds__(256)
void normmod_kernel(const float* __restrict__ x, const float* __restrict__ nscale,
                    const float* __restrict__ mods, int shoff, int scoff,
                    int modstride, __bf16* __restrict__ out, int Sn, int Dn)
{
  int row = blockIdx.x;
  int b = row / Sn;
  const float* xr = x + (size_t)row * Dn;
  float xv[4];
  float ss = 0.f;
  for (int i = 0; i < 4; ++i) {
    int d = i * 256 + threadIdx.x;
    xv[i] = xr[d];
    ss += xv[i] * xv[i];
  }
  for (int off = 16; off; off >>= 1) ss += __shfl_xor(ss, off, 32);
  __shared__ float red[8];
  if ((threadIdx.x & 31) == 0) red[threadIdx.x >> 5] = ss;
  __syncthreads();
  float tot = 0.f;
  for (int i = 0; i < 8; ++i) tot += red[i];
  float rn = rsqrtf(tot / Dn + 1e-6f);
  for (int i = 0; i < 4; ++i) {
    int d = i * 256 + threadIdx.x;
    float sh = mods[(size_t)b * modstride + shoff + d];
    float sc = mods[(size_t)b * modstride + scoff + d];
    out[(size_t)row * Dn + d] = (__bf16)((xv[i] * rn * nscale[d]) * (1.f + sc) + sh);
  }
}

__global__ void silu_kernel(const float* __restrict__ c, float* __restrict__ o, int n) {
  int i = blockIdx.x * 256 + threadIdx.x;
  if (i < n) o[i] = silu_f(c[i]);
}

// mods = silu(c) @ w_ada + b_ada   (B=8 rows; one thread per output)
__global__ void ada_kernel(const float* __restrict__ sc, const float* __restrict__ w,
                           const float* __restrict__ bias, float* __restrict__ mods,
                           int Dn, int N6) {
  int id = blockIdx.x * 256 + threadIdx.x;
  int b = id / N6, j = id % N6;
  const float* cr = sc + (size_t)b * Dn;
  float a = bias[j];
  for (int kk = 0; kk < Dn; ++kk) a += cr[kk] * w[(size_t)kk * N6 + j];
  mods[id] = a;
}

// w[K][N] fp32 -> wt[N][K] bf16
__global__ void convt_kernel(const float* __restrict__ w, __bf16* __restrict__ wt,
                             int K, int N) {
  long long idx = (long long)blockIdx.x * 256 + threadIdx.x;
  if (idx >= (long long)K * N) return;
  int kk = (int)(idx / N), n = (int)(idx % N);
  wt[(size_t)n * K + kk] = (__bf16)w[idx];
}

// gate = silu(x1) * x2 from x12[M][2*inner] bf16
__global__ void swiglu_kernel(const __bf16* __restrict__ x12,
                              __bf16* __restrict__ gate, long long total, int inner) {
  long long idx = (long long)blockIdx.x * 256 + threadIdx.x;
  if (idx >= total) return;
  long long r = idx / inner;
  int i = (int)(idx % inner);
  float a = (float)x12[r * (2LL * inner) + i];
  float bb = (float)x12[r * (2LL * inner) + inner + i];
  gate[idx] = (__bf16)(silu_f(a) * bb);
}

// ---------------------------------------------------------------------------
extern "C" void kernel_launch(void* const* d_in, const int* in_sizes, int n_in,
                              void* d_out, int out_size, void* d_ws, size_t ws_size,
                              hipStream_t stream)
{
  (void)in_sizes; (void)n_in; (void)out_size; (void)ws_size;
  const int B = 8, S = 1024, D = 1024, H = 16, INNER = 2730;
  const int N6 = 6 * D;

  const float* x     = (const float*)d_in[0];
  const float* c     = (const float*)d_in[1];
  const float* rc    = (const float*)d_in[2];
  const float* rs    = (const float*)d_in[3];
  const float* n1s   = (const float*)d_in[4];
  const float* n2s   = (const float*)d_in[5];
  const float* qns   = (const float*)d_in[6];
  const float* kns   = (const float*)d_in[7];
  const float* wqkv  = (const float*)d_in[8];
  const float* bqkv  = (const float*)d_in[9];
  const float* wproj = (const float*)d_in[10];
  const float* bproj = (const float*)d_in[11];
  const float* w12   = (const float*)d_in[12];
  const float* b12   = (const float*)d_in[13];
  const float* w3    = (const float*)d_in[14];
  const float* b3    = (const float*)d_in[15];
  const float* wada  = (const float*)d_in[16];
  const float* bada  = (const float*)d_in[17];
  float* out = (float*)d_out;

  char* ws = (char*)d_ws;
  size_t off = 0;
  auto alloc = [&](size_t bytes) {
    size_t o = off;
    off = (off + bytes + 255) & ~(size_t)255;
    return o;
  };
  float*  mods   = (float*)(ws + alloc((size_t)B * N6 * 4));
  float*  siluc  = (float*)(ws + alloc((size_t)B * D * 4));
  __bf16* wqkvT  = (__bf16*)(ws + alloc((size_t)3 * D * D * 2));
  __bf16* wprojT = (__bf16*)(ws + alloc((size_t)D * D * 2));
  __bf16* w12T   = (__bf16*)(ws + alloc((size_t)2 * INNER * D * 2));
  __bf16* w3T    = (__bf16*)(ws + alloc((size_t)D * INNER * 2));
  __bf16* hbuf   = (__bf16*)(ws + alloc((size_t)B * S * D * 2));
  size_t qkv_off = alloc((size_t)B * S * 3 * D * 4);
  float*  qkv    = (float*)(ws + qkv_off);
  size_t q_off   = alloc((size_t)B * H * S * HD * 2);
  __bf16* qb     = (__bf16*)(ws + q_off);
  __bf16* kb     = (__bf16*)(ws + alloc((size_t)B * H * S * HD * 2));
  __bf16* vtb    = (__bf16*)(ws + alloc((size_t)B * H * S * HD * 2));
  __bf16* ob     = (__bf16*)(ws + alloc((size_t)B * S * D * 2));
  float*  x2     = (float*)(ws + alloc((size_t)B * S * D * 4));
  __bf16* h2     = hbuf;                       // alias (hbuf dead after QKV GEMM)
  __bf16* x12    = (__bf16*)(ws + qkv_off);    // alias (qkv dead after rope; 89MB<=96MB)
  __bf16* gate   = (__bf16*)(ws + q_off);      // alias (q/k/vt dead after attn; 45MB<=48MB)

  const int tpb = 256;
  // 1) weight bf16 transposes
  convt_kernel<<<(D * 3 * D + tpb - 1) / tpb, tpb, 0, stream>>>(wqkv, wqkvT, D, 3 * D);
  convt_kernel<<<(D * D + tpb - 1) / tpb, tpb, 0, stream>>>(wproj, wprojT, D, D);
  convt_kernel<<<(D * 2 * INNER + tpb - 1) / tpb, tpb, 0, stream>>>(w12, w12T, D, 2 * INNER);
  convt_kernel<<<(INNER * D + tpb - 1) / tpb, tpb, 0, stream>>>(w3, w3T, INNER, D);
  // 2) adaLN mods
  silu_kernel<<<(B * D + tpb - 1) / tpb, tpb, 0, stream>>>(c, siluc, B * D);
  ada_kernel<<<B * N6 / tpb, tpb, 0, stream>>>(siluc, wada, bada, mods, D, N6);
  // 3) norm1 + modulate -> h (bf16)
  normmod_kernel<<<B * S, tpb, 0, stream>>>(x, n1s, mods, 0, D, N6, hbuf, S, D);
  // 4) QKV GEMM (fp32 out)
  dim3 g1(3 * D / 128, B * S / 128);
  gemm_kernel<<<g1, tpb, 0, stream>>>(hbuf, wqkvT, bqkv, B * S, 3 * D, D, 0,
                                      qkv, nullptr, nullptr, nullptr, 0, 0, S);
  // 5) head RMSNorm + RoPE + V transpose
  rope_kernel<<<B * S * H / 8, tpb, 0, stream>>>(qkv, rc, rs, qns, kns,
                                                 qb, kb, vtb, B, S, H);
  // 6) attention
  dim3 ga(S / 64, B * H);
  attn_kernel<<<ga, 128, 0, stream>>>(qb, kb, vtb, ob, S, H);
  // 7) proj GEMM + residual (x2 = x + g_msa * o)
  dim3 g2(D / 128, B * S / 128);
  gemm_kernel<<<g2, tpb, 0, stream>>>(ob, wprojT, bproj, B * S, D, D, 2,
                                      x2, nullptr, x, mods, 2 * D, N6, S);
  // 8) norm2 + modulate -> h2
  normmod_kernel<<<B * S, tpb, 0, stream>>>(x2, n2s, mods, 3 * D, 4 * D, N6, h2, S, D);
  // 9) w12 GEMM (bf16 out, N=5460 masked)
  dim3 g3((2 * INNER + 127) / 128, B * S / 128);
  gemm_kernel<<<g3, tpb, 0, stream>>>(h2, w12T, b12, B * S, 2 * INNER, D, 1,
                                      nullptr, x12, nullptr, nullptr, 0, 0, S);
  // 10) SwiGLU
  long long gt = (long long)B * S * INNER;
  swiglu_kernel<<<(unsigned)((gt + tpb - 1) / tpb), tpb, 0, stream>>>(x12, gate, gt, INNER);
  // 11) w3 GEMM (K=2730 guarded) + residual (out = x2 + g_mlp * mlp)
  gemm_kernel<<<g2, tpb, 0, stream>>>(gate, w3T, b3, B * S, D, INNER, 2,
                                      out, nullptr, x2, mods, 5 * D, N6, S);
}
